// MultiHeadSelfAttention_46995532153203
// MI455X (gfx1250) — matrix operand order repaired and checked by reading, not compiled, not run on hardware
//
#include <hip/hip_runtime.h>
#include <hip/hip_bf16.h>
#include <stdint.h>

#define D_MODEL 1024
#define NHEAD   16
#define DKH     64
#define BATCH   2
#define SEQ     2048

typedef __attribute__((ext_vector_type(16))) __bf16 v16bf;
typedef __attribute__((ext_vector_type(8)))  __bf16 v8bf;
typedef __attribute__((ext_vector_type(8)))  float  v8f;
typedef __attribute__((ext_vector_type(4))) unsigned int u32x4;
typedef __attribute__((ext_vector_type(8))) int i32x8;
typedef __attribute__((ext_vector_type(4))) int i32x4;

#if defined(__gfx1250__) && __has_builtin(__builtin_amdgcn_tensor_load_to_lds)
#define HAVE_TDM 1
#else
#define HAVE_TDM 0
#endif

#if HAVE_TDM
// TDM 2D tile load: global row-major tile (tile_w elems/row, tile_h rows,
// row_stride elems between rows) -> LDS, with pad_amt DWORDs inserted every
// pad_int DWORDs (codes per D# spec: interval 0=2DW..4=32DW, amount 0=1DW..).
// D# built per CDNA5 ISA ch.8: group0 = {flags, lds_addr, global_addr, type=2},
// group1 = {mask/size/pad, tensor dims, tile dims, strides}. 2D: groups 2/3=0.
// This toolchain's builtin is the 6-arg form: (g0, g1, g2, g3, g_ext, cpol).
__device__ inline void tdm_load_2d(const void* gptr, void* lds_ptr,
                                   uint32_t tile_w, uint32_t tile_h,
                                   uint32_t row_stride, uint32_t pad_int_code,
                                   uint32_t pad_amt_code) {
  uint64_t ga = (uint64_t)(uintptr_t)gptr;
  u32x4 g0;
  g0[0] = 1u;                                  // count=1, user descriptor
  g0[1] = (uint32_t)(uintptr_t)lds_ptr;        // lds_addr (LDS byte offset)
  g0[2] = (uint32_t)ga;                        // global_addr[31:0]
  g0[3] = (uint32_t)((ga >> 32) & 0x01FFFFFFu) | (2u << 30);  // [56:32], type=2
  i32x8 g1;
  g1[0] = (int)((1u << 16) |                   // data_size = 2 bytes
                (1u << 20) |                   // pad_enable
                (pad_int_code << 22) | (pad_amt_code << 25));
  g1[1] = (int)((tile_w & 0xFFFFu) << 16);     // tensor_dim0 (bits 79:48)
  g1[2] = (int)(((tile_w >> 16) & 0xFFFFu) | ((tile_h & 0xFFFFu) << 16)); // dim1
  g1[3] = (int)(((tile_h >> 16) & 0xFFFFu) | ((tile_w & 0xFFFFu) << 16)); // tile_dim0
  g1[4] = (int)(tile_h & 0xFFFFu);             // tile_dim1 ; tile_dim2 = 0
  g1[5] = (int)row_stride;                     // tensor_dim0_stride[31:0]
  g1[6] = 0;
  g1[7] = 0;
  i32x4 z4 = {0, 0, 0, 0};
  i32x8 z8 = {0, 0, 0, 0, 0, 0, 0, 0};
  __builtin_amdgcn_tensor_load_to_lds(g0, g1, z4, z4, z8, 0);
}
#endif

// ---------------------------------------------------------------- conversion
__global__ void cvt_f32_to_bf16(const float* __restrict__ in,
                                __bf16* __restrict__ out, int n) {
  int i = blockIdx.x * blockDim.x + threadIdx.x;
  int stride = gridDim.x * blockDim.x;
  for (; i < n; i += stride) out[i] = (__bf16)in[i];
}

// ---------------------------------------------------------------- GEMM
// C[M,N] = A[M,K] @ B[K,N] + bias ; bf16 in, f32 accumulate.
// Block tile 128x64, K-tile 32, double-buffered LDS; 8 waves, 32x32 per wave.
// A tile staged by TDM (async, TENSORcnt, wave 0 only via scalar branch);
// B staged manually (transposed into LDS).
template <bool OUT_BF16>
__global__ __launch_bounds__(256) void gemm_bf16_wmma(
    const __bf16* __restrict__ A, const __bf16* __restrict__ B,
    const float* __restrict__ bias, void* __restrict__ Cout,
    int M, int N, int K) {
  constexpr int BM = 128, BN = 64, BK = 32;
  constexpr int LDA = BK + 8;   // 40 elems = 80B (pad codes 3/3 -> 16DW + 4DW)
  constexpr int LDB = BK + 8;
  __shared__ __align__(16) __bf16 As[2][BM * LDA];
  __shared__ __align__(16) __bf16 Bt[2][BN * LDB];

  const int tid  = threadIdx.x;
  const int lane = tid & 31;
  const int half = lane >> 4;
  const int r    = lane & 15;
  // scalar (SGPR) wave id -> uniform branch around TDM, no EXEC masking
  const int swave = __builtin_amdgcn_readfirstlane(tid) >> 5;
  const int wm   = swave & 3;
  const int wn   = swave >> 2;
  const int block_m = blockIdx.y * BM;
  const int block_n = blockIdx.x * BN;

  const int arow = tid >> 1, ach = (tid & 1) * 16;  // A stage: 128 rows x 32
  const int bkk  = tid >> 3, bn0 = (tid & 7) * 8;   // B stage: 32 rows x 64

  v8f c[2][2] = {};

  // ---- prologue: stage tile 0 into buffer 0
#if HAVE_TDM
  if (swave == 0)
    tdm_load_2d(A + (size_t)block_m * K, &As[0][0], BK, BM, (uint32_t)K, 3u, 3u);
#else
  {
    const __bf16* g = A + (size_t)(block_m + arow) * K + ach;
    v8bf a0 = *(const v8bf*)g;
    v8bf a1 = *(const v8bf*)(g + 8);
    *(v8bf*)&As[0][arow * LDA + ach]     = a0;
    *(v8bf*)&As[0][arow * LDA + ach + 8] = a1;
  }
#endif
  {
    const __bf16* g = B + (size_t)bkk * N + block_n + bn0;
    v8bf bv = *(const v8bf*)g;
#pragma unroll
    for (int e = 0; e < 8; ++e) Bt[0][(bn0 + e) * LDB + bkk] = bv[e];
  }
#if HAVE_TDM
  if (swave == 0) __builtin_amdgcn_s_wait_tensorcnt(0);
#endif
  __syncthreads();

  for (int kt = 0; kt < K; kt += BK) {
    const int buf = (kt / BK) & 1;
    const bool has_next = (kt + BK) < K;

    // ---- prefetch next tile while current tile computes
    v8bf bv_n = {};
#if !HAVE_TDM
    v8bf a0_n = {}, a1_n = {};
#endif
    if (has_next) {
#if HAVE_TDM
      if (swave == 0)
        tdm_load_2d(A + (size_t)block_m * K + kt + BK, &As[buf ^ 1][0], BK, BM,
                    (uint32_t)K, 3u, 3u);
#else
      const __bf16* ga = A + (size_t)(block_m + arow) * K + kt + BK + ach;
      a0_n = *(const v8bf*)ga;
      a1_n = *(const v8bf*)(ga + 8);
#endif
      const __bf16* gb = B + (size_t)(kt + BK + bkk) * N + block_n + bn0;
      bv_n = *(const v8bf*)gb;
    }

    // ---- fragments (A and B both in contiguous k = 16*half + s order) + 4 WMMAs
    v16bf af[2], bfx[2];
#pragma unroll
    for (int mi = 0; mi < 2; ++mi) {
      const __bf16* p = &As[buf][(wm * 32 + mi * 16 + r) * LDA];
      v8bf lo = *(const v8bf*)(p + half * 16);
      v8bf hi = *(const v8bf*)(p + half * 16 + 8);
#pragma unroll
      for (int e = 0; e < 8; ++e) { af[mi][e] = lo[e]; af[mi][8 + e] = hi[e]; }
    }
#pragma unroll
    for (int ni = 0; ni < 2; ++ni) {
      const __bf16* p = &Bt[buf][(wn * 32 + ni * 16 + r) * LDB + half * 16];
      v8bf lo = *(const v8bf*)p;
      v8bf hi = *(const v8bf*)(p + 8);
#pragma unroll
      for (int e = 0; e < 8; ++e) { bfx[ni][e] = lo[e]; bfx[ni][8 + e] = hi[e]; }
    }
#pragma unroll
    for (int mi = 0; mi < 2; ++mi)
#pragma unroll
      for (int ni = 0; ni < 2; ++ni)
        c[mi][ni] = __builtin_amdgcn_wmma_f32_16x16x32_bf16(
            false, af[mi], false, bfx[ni], (short)0, c[mi][ni], false, false);

    // ---- commit prefetched tile into the other buffer
    if (has_next) {
#if !HAVE_TDM
      *(v8bf*)&As[buf ^ 1][arow * LDA + ach]     = a0_n;
      *(v8bf*)&As[buf ^ 1][arow * LDA + ach + 8] = a1_n;
#endif
#pragma unroll
      for (int e = 0; e < 8; ++e) Bt[buf ^ 1][(bn0 + e) * LDB + bkk] = bv_n[e];
    }
#if HAVE_TDM
    if (swave == 0) __builtin_amdgcn_s_wait_tensorcnt(0);
#endif
    __syncthreads();
  }

  // ---- store: C layout row = vgpr_i + half*8, col = lane%16
#pragma unroll
  for (int mi = 0; mi < 2; ++mi)
#pragma unroll
    for (int ni = 0; ni < 2; ++ni) {
      int col = block_n + wn * 32 + ni * 16 + r;
      float bv = bias ? bias[col] : 0.0f;
#pragma unroll
      for (int i = 0; i < 8; ++i) {
        int row = block_m + wm * 32 + mi * 16 + half * 8 + i;
        float v = c[mi][ni][i] + bv;
        if (OUT_BF16) ((__bf16*)Cout)[(size_t)row * N + col] = (__bf16)v;
        else          ((float*)Cout)[(size_t)row * N + col]  = v;
      }
    }
}

// ---------------------------------------------------------------- flash attention
// qkv: [B, L, 3, H, DKH] bf16 ; mask: [B, L] i32 ; ctx: [B, L, H, DKH] bf16
// 128 threads = 4 waves; wave owns 16 query rows; key tiles of 32.
// K tile staged by TDM (row stride 3*D_MODEL); V staged transposed manually.
__global__ __launch_bounds__(128) void flash_attn_wmma(
    const __bf16* __restrict__ qkv, const int* __restrict__ mask,
    __bf16* __restrict__ ctx) {
  constexpr int KT  = 32;
  constexpr int LDK = DKH + 8;  // 72 elems = 144B (pad codes 4/3 -> 32DW + 4DW)
  constexpr int LDV = KT + 8;
  constexpr int LDP = KT + 8;
  __shared__ __align__(16) __bf16 Ks[KT * LDK];
  __shared__ __align__(16) __bf16 Vt[DKH * LDV];
  __shared__ __align__(16) __bf16 Ps[4][16 * LDP];

  const int tid  = threadIdx.x;
  const int lane = tid & 31;
  const int half = lane >> 4;
  const int r    = lane & 15;
  const int swave = __builtin_amdgcn_readfirstlane(tid) >> 5;  // uniform wave id
  const int h    = blockIdx.y;
  const int b    = blockIdx.z;
  const int q0   = blockIdx.x * 64 + swave * 16;
  const float scale = 0.125f;  // 1/sqrt(64)

  // Q fragments (A layout), dk=64 -> two K32 chunks, loaded once
  v16bf qa0, qa1;
  {
    const __bf16* qp =
        qkv + ((size_t)(b * SEQ + q0 + r) * 3 + 0) * D_MODEL + h * DKH;
    v8bf x0 = *(const v8bf*)(qp + half * 16);
    v8bf x1 = *(const v8bf*)(qp + half * 16 + 8);
    v8bf x2 = *(const v8bf*)(qp + 32 + half * 16);
    v8bf x3 = *(const v8bf*)(qp + 32 + half * 16 + 8);
#pragma unroll
    for (int e = 0; e < 8; ++e) {
      qa0[e] = x0[e]; qa0[8 + e] = x1[e];
      qa1[e] = x2[e]; qa1[8 + e] = x3[e];
    }
  }

  float mrow[8], lrow[8];
  v8f o[4] = {};
#pragma unroll
  for (int i = 0; i < 8; ++i) { mrow[i] = -1e30f; lrow[i] = 0.0f; }

  for (int key0 = 0; key0 < SEQ; key0 += KT) {
    // --- stage K (TDM, row-major padded) and V (manual, transposed)
#if HAVE_TDM
    if (swave == 0)
      tdm_load_2d(qkv + ((size_t)(b * SEQ + key0) * 3 + 1) * D_MODEL + h * DKH,
                  &Ks[0], DKH, KT, 3 * D_MODEL, 4u, 3u);
#endif
    {
      int key = tid >> 2;
      int ch  = (tid & 3) * 16;
#if !HAVE_TDM
      const __bf16* kg =
          qkv + ((size_t)(b * SEQ + key0 + key) * 3 + 1) * D_MODEL + h * DKH + ch;
      v8bf k0v = *(const v8bf*)kg;
      v8bf k1v = *(const v8bf*)(kg + 8);
      *(v8bf*)&Ks[key * LDK + ch]     = k0v;
      *(v8bf*)&Ks[key * LDK + ch + 8] = k1v;
#endif
      const __bf16* vg =
          qkv + ((size_t)(b * SEQ + key0 + key) * 3 + 2) * D_MODEL + h * DKH + ch;
      v8bf v0v = *(const v8bf*)vg;
      v8bf v1v = *(const v8bf*)(vg + 8);
#pragma unroll
      for (int e = 0; e < 8; ++e) {
        Vt[(ch + e) * LDV + key]     = v0v[e];
        Vt[(ch + 8 + e) * LDV + key] = v1v[e];
      }
    }
#if HAVE_TDM
    if (swave == 0) __builtin_amdgcn_s_wait_tensorcnt(0);
#endif
    __syncthreads();

    // --- S = Q @ K^T : two 16x16 column tiles
    v8f s[2] = {};
#pragma unroll
    for (int nt = 0; nt < 2; ++nt) {
      v16bf kb0, kb1;
      const __bf16* p = &Ks[(nt * 16 + r) * LDK];
      v8bf c0 = *(const v8bf*)(p + half * 16);
      v8bf c1 = *(const v8bf*)(p + half * 16 + 8);
      v8bf c2 = *(const v8bf*)(p + 32 + half * 16);
      v8bf c3 = *(const v8bf*)(p + 32 + half * 16 + 8);
#pragma unroll
      for (int e = 0; e < 8; ++e) {
        kb0[e] = c0[e]; kb0[8 + e] = c1[e];
        kb1[e] = c2[e]; kb1[8 + e] = c3[e];
      }
      s[nt] = __builtin_amdgcn_wmma_f32_16x16x32_bf16(
          false, qa0, false, kb0, (short)0, s[nt], false, false);
      s[nt] = __builtin_amdgcn_wmma_f32_16x16x32_bf16(
          false, qa1, false, kb1, (short)0, s[nt], false, false);
    }

    const bool mok0 = mask[b * SEQ + key0 + r] != 0;
    const bool mok1 = mask[b * SEQ + key0 + 16 + r] != 0;

    // --- online softmax; row stats per lane (row = i + half*8)
    float p0[8], p1[8], alpha[8];
#pragma unroll
    for (int i = 0; i < 8; ++i) {
      float s0 = mok0 ? s[0][i] * scale : -1e9f;
      float s1 = mok1 ? s[1][i] * scale : -1e9f;
      float vmax = fmaxf(s0, s1);
#pragma unroll
      for (int mm = 1; mm < 16; mm <<= 1)
        vmax = fmaxf(vmax, __shfl_xor(vmax, mm, 16));
      float mnew = fmaxf(mrow[i], vmax);
      float a = __expf(mrow[i] - mnew);
      p0[i] = __expf(s0 - mnew);
      p1[i] = __expf(s1 - mnew);
      float ps = p0[i] + p1[i];
#pragma unroll
      for (int mm = 1; mm < 16; mm <<= 1) ps += __shfl_xor(ps, mm, 16);
      lrow[i] = lrow[i] * a + ps;
      mrow[i] = mnew;
      alpha[i] = a;
    }
#pragma unroll
    for (int j = 0; j < 4; ++j)
#pragma unroll
      for (int i = 0; i < 8; ++i) o[j][i] *= alpha[i];

    // --- re-layout P (C layout -> A layout) via per-wave LDS
    __bf16* psb = &Ps[swave][0];
#pragma unroll
    for (int i = 0; i < 8; ++i) {
      psb[(half * 8 + i) * LDP + r]      = (__bf16)p0[i];
      psb[(half * 8 + i) * LDP + 16 + r] = (__bf16)p1[i];
    }
    v16bf pa;
    {
      const __bf16* p = &psb[r * LDP];
      v8bf lo = *(const v8bf*)(p + half * 16);
      v8bf hi = *(const v8bf*)(p + half * 16 + 8);
#pragma unroll
      for (int e = 0; e < 8; ++e) { pa[e] = lo[e]; pa[8 + e] = hi[e]; }
    }

    // --- O += P @ V  (16x32 @ 32x64 -> four 16x16 tiles)
#pragma unroll
    for (int j = 0; j < 4; ++j) {
      v16bf vb;
      const __bf16* p = &Vt[(j * 16 + r) * LDV + half * 16];
      v8bf lo = *(const v8bf*)p;
      v8bf hi = *(const v8bf*)(p + 8);
#pragma unroll
      for (int e = 0; e < 8; ++e) { vb[e] = lo[e]; vb[8 + e] = hi[e]; }
      o[j] = __builtin_amdgcn_wmma_f32_16x16x32_bf16(
          false, pa, false, vb, (short)0, o[j], false, false);
    }
    __syncthreads();
  }

  // --- epilogue: divide by softmax denominator, store ctx[B,L,H,dk]
#pragma unroll
  for (int i = 0; i < 8; ++i) {
    float inv = 1.0f / lrow[i];
    int row = q0 + half * 8 + i;
    __bf16* cp = ctx + (size_t)(b * SEQ + row) * D_MODEL + h * DKH;
#pragma unroll
    for (int j = 0; j < 4; ++j) cp[j * 16 + r] = (__bf16)(o[j][i] * inv);
  }
}

// ---------------------------------------------------------------- launch
extern "C" void kernel_launch(void* const* d_in, const int* in_sizes, int n_in,
                              void* d_out, int out_size, void* d_ws,
                              size_t ws_size, hipStream_t stream) {
  (void)in_sizes; (void)n_in; (void)out_size; (void)ws_size;
  const float* x     = (const float*)d_in[0];
  const int*   maskp = (const int*)d_in[1];
  const float* W_qkv = (const float*)d_in[2];
  const float* b_qkv = (const float*)d_in[3];
  const float* W_out = (const float*)d_in[4];
  const float* b_out = (const float*)d_in[5];

  const int M = BATCH * SEQ;  // 4096
  char* ws = (char*)d_ws;
  size_t off = 0;
  auto alloc = [&](size_t bytes) {
    void* p = ws + off;
    off += (bytes + 255) & ~(size_t)255;
    return p;
  };
  __bf16* xb    = (__bf16*)alloc((size_t)M * D_MODEL * 2);
  __bf16* wqkvb = (__bf16*)alloc((size_t)D_MODEL * 3 * D_MODEL * 2);
  __bf16* woutb = (__bf16*)alloc((size_t)D_MODEL * D_MODEL * 2);
  __bf16* qkvb  = (__bf16*)alloc((size_t)M * 3 * D_MODEL * 2);
  __bf16* ctxb  = (__bf16*)alloc((size_t)M * D_MODEL * 2);

  {
    int n0 = M * D_MODEL, n1 = D_MODEL * 3 * D_MODEL, n2 = D_MODEL * D_MODEL;
    cvt_f32_to_bf16<<<dim3((n0 + 2047) / 2048), dim3(256), 0, stream>>>(x, xb, n0);
    cvt_f32_to_bf16<<<dim3((n1 + 2047) / 2048), dim3(256), 0, stream>>>(W_qkv, wqkvb, n1);
    cvt_f32_to_bf16<<<dim3((n2 + 2047) / 2048), dim3(256), 0, stream>>>(W_out, woutb, n2);
  }
  gemm_bf16_wmma<true><<<dim3(3 * D_MODEL / 64, M / 128), dim3(256), 0, stream>>>(
      xb, wqkvb, b_qkv, qkvb, M, 3 * D_MODEL, D_MODEL);
  flash_attn_wmma<<<dim3(SEQ / 64, NHEAD, BATCH), dim3(128), 0, stream>>>(
      qkvb, maskp, ctxb);
  gemm_bf16_wmma<false><<<dim3(D_MODEL / 64, M / 128), dim3(256), 0, stream>>>(
      ctxb, woutb, b_out, (float*)d_out, M, D_MODEL, D_MODEL);
}
